// EQGATLocalGNN_31344671326802
// MI455X (gfx1250) — matrix operand order repaired and checked by reading, not compiled, not run on hardware
//
#include <hip/hip_runtime.h>
#include <hip/hip_bf16.h>
#include <cstdint>
#include <cstddef>

#define SI 128
#define VI 32
#define EDGE_DIM 16
#define NLAYERS 5
#define NNODES 16384
#define NEDGES 262144
#define NGRAPHS 256
#define EPSF 1e-6f

typedef __attribute__((ext_vector_type(16))) __bf16   v16bf;
typedef __attribute__((ext_vector_type(8)))  float    v8f;
typedef __attribute__((ext_vector_type(4)))  unsigned v4u;
typedef __attribute__((ext_vector_type(8)))  unsigned v8u;

#define WMMA_BF16(A, B, C) \
  __builtin_amdgcn_wmma_f32_16x16x32_bf16(false, (A), false, (B), (short)0, (C), false, false)

__device__ __forceinline__ __bf16 f2bf(float f) {
  unsigned u = __builtin_bit_cast(unsigned, f);
  unsigned r = (u + 0x7FFFu + ((u >> 16) & 1u)) >> 16;
  return __builtin_bit_cast(__bf16, (unsigned short)r);
}
__device__ __forceinline__ unsigned pack2bf(float a, float b) {
  unsigned ua = __builtin_bit_cast(unsigned, a);
  unsigned ub = __builtin_bit_cast(unsigned, b);
  unsigned ra = (ua + 0x7FFFu + ((ua >> 16) & 1u)) >> 16;
  unsigned rb = (ub + 0x7FFFu + ((ub >> 16) & 1u)) >> 16;
  return (ra & 0xFFFFu) | (rb << 16);
}
__device__ __forceinline__ float siluf(float x) { return x / (1.f + __expf(-x)); }
__device__ __forceinline__ float sigmf(float x) { return 1.f / (1.f + __expf(-x)); }

__device__ __forceinline__ void atomicMaxF(float* addr, float val) {
  if (val >= 0.f) atomicMax((int*)addr, __float_as_int(val));
  else            atomicMin((unsigned int*)addr, __float_as_uint(val));
}

// A fragment: lane holds row nn; K pattern = [k0+half*16 .. +15] (contiguous)
__device__ __forceinline__ v16bf load_afrag(const __bf16* row, int k0, int half) {
  v4u lo = *(const v4u*)(row + k0 + half * 16);
  v4u hi = *(const v4u*)(row + k0 + half * 16 + 8);
  v8u u = __builtin_shufflevector(lo, hi, 0, 1, 2, 3, 4, 5, 6, 7);
  return __builtin_bit_cast(v16bf, u);
}
// B fragment: 16 contiguous bf16 (32B, 32B-aligned)
__device__ __forceinline__ v16bf load_bfrag(const __bf16* p) {
  v8u u = *(const v8u*)p;
  return __builtin_bit_cast(v16bf, u);
}

// ---------------- utility kernels ----------------
__global__ void k_fill(float* p, float v, int n) {
  int i = blockIdx.x * 256 + threadIdx.x; if (i < n) p[i] = v;
}
__global__ void k_fill_bits(unsigned* p, unsigned v, int n) {
  int i = blockIdx.x * 256 + threadIdx.x; if (i < n) p[i] = v;
}
__global__ void k_copy(float* dst, const float* src, int n) {
  int i = blockIdx.x * 256 + threadIdx.x; if (i < n) dst[i] = src[i];
}
__global__ void k_add(float* dst, const float* src, int n) {
  int i = blockIdx.x * 256 + threadIdx.x; if (i < n) dst[i] += src[i];
}

// W[K][N] f32 row-major -> Wt[Npad][Kpad] bf16 (transposed, zero padded)
__global__ void k_convert_wt(const float* W, __bf16* Wt, int K, int N, int Kpad, int Npad) {
  int idx = blockIdx.x * 256 + threadIdx.x;
  if (idx >= Kpad * Npad) return;
  int n = idx / Kpad, k = idx % Kpad;
  float v = (k < K && n < N) ? W[k * N + n] : 0.f;
  Wt[n * Kpad + k] = f2bf(v);
}

// ---------------- graph layernorm stats ----------------
__global__ void k_node_stats1(const float* s, const float* v, const int* batch,
                              float* gsum_s, float* gsum_v, float* gcnt) {
  __shared__ float red[128];
  int n = blockIdx.x, t = threadIdx.x;
  red[t] = s[n * SI + t];
  __syncthreads();
  for (int off = 64; off; off >>= 1) { if (t < off) red[t] += red[t + off]; __syncthreads(); }
  float tot_s = red[0];
  __syncthreads();
  float q = 0.f;
  if (t < 96) { float a = v[n * 96 + t]; q = a * a; }
  red[t] = q;
  __syncthreads();
  for (int off = 64; off; off >>= 1) { if (t < off) red[t] += red[t + off]; __syncthreads(); }
  if (t == 0) {
    int b = batch[n];
    atomicAdd(&gsum_s[b], tot_s * (1.f / SI));
    atomicAdd(&gsum_v[b], red[0] * (1.f / VI));
    atomicAdd(&gcnt[b], 1.f);
  }
}
__global__ void k_graph_mu(const float* gsum_s, const float* gcnt, float* mu) {
  int g = threadIdx.x;
  mu[g] = gsum_s[g] / fmaxf(gcnt[g], 1.f);
}
__global__ void k_node_stats2(const float* s, const int* batch, const float* mu, float* gsum_var) {
  __shared__ float red[128];
  int n = blockIdx.x, t = threadIdx.x;
  float sc = s[n * SI + t] - mu[batch[n]];
  red[t] = sc * sc;
  __syncthreads();
  for (int off = 64; off; off >>= 1) { if (t < off) red[t] += red[t + off]; __syncthreads(); }
  if (t == 0) atomicAdd(&gsum_var[batch[n]], red[0] * (1.f / SI));
}
__global__ void k_graph_fin(const float* gsum_var, const float* gsum_v, const float* gcnt,
                            float* rstd_s, float* rstd_v) {
  int g = threadIdx.x;
  float c = fmaxf(gcnt[g], 1.f);
  rstd_s[g] = rsqrtf(gsum_var[g] / c + EPSF);
  rstd_v[g] = rsqrtf(gsum_v[g] / c + EPSF);
}
__global__ void k_normalize(float* s, float* v, const int* batch, const float* mu,
                            const float* rstd_s, const float* rstd_v,
                            const float* lnw, const float* lnb) {
  int i = blockIdx.x * 256 + threadIdx.x;
  if (i >= NNODES * (SI + 96)) return;
  int n = i / (SI + 96), c = i % (SI + 96);
  int b = batch[n];
  if (c < SI) {
    float sc = s[n * SI + c] - mu[b];
    s[n * SI + c] = sc * rstd_s[b] * lnw[c] + lnb[c];
  } else {
    v[n * 96 + (c - SI)] *= rstd_v[b];
  }
}

// ---------------- generic WMMA GEMM ----------------
// Y[M,N] = act(X[M,K] @ W + bias); W transposed bf16 Wt[Npad][Kpad]
// K <= 160 and a multiple of 32.  128 threads = 4 waves; whole 16xK panel staged once.
#define KMAXP 168   // 160 + 8 pad (bank-conflict-free, keeps 16B alignment)
__global__ void k_gemm(const float* __restrict__ X, const __bf16* __restrict__ Wt,
                       const float* __restrict__ bias, float* __restrict__ Y,
                       int M, int N, int K, int ldx, int ldy, int Kpad,
                       int act, int addout) {
  __shared__ __bf16 xs[16][KMAXP];
  int tid = threadIdx.x;
  int w = tid >> 5, lane = tid & 31;
  int half = lane >> 4, nn = lane & 15;
  int m0 = blockIdx.y * 16;
  int n0 = blockIdx.x * 64 + w * 16;
  // stage 16 x K panel: 16 rows x 8 threads, float4 loads, packed bf16 dword stores
  {
    int r = tid >> 3, t8 = tid & 7;
    const float4* src = (const float4*)(X + (m0 + r) * ldx);
    int kv = K >> 2;
    for (int c4 = t8; c4 < kv; c4 += 8) {
      float4 v = src[c4];
      unsigned* dp = (unsigned*)&xs[r][c4 << 2];
      dp[0] = pack2bf(v.x, v.y);
      dp[1] = pack2bf(v.z, v.w);
    }
  }
  __syncthreads();
  v8f acc = {};
  int ksteps = K >> 5;
  const __bf16* wrow = Wt + (n0 + nn) * Kpad + half * 16;
  for (int ks = 0; ks < ksteps; ++ks) {
    int k0 = ks << 5;
    v16bf a = load_afrag(&xs[nn][0], k0, half);
    v16bf b = load_bfrag(wrow + k0);
    acc = WMMA_BF16(a, b, acc);
  }
  int col = n0 + nn;
  if (col < N) {
    float bv = bias ? bias[col] : 0.f;
#pragma unroll
    for (int r = 0; r < 8; ++r) {
      int row = m0 + half * 8 + r;
      float val = acc[r] + bv;
      if (act == 1) val = siluf(val);
      float* yp = Y + row * ldy + col;
      if (addout) *yp += val; else *yp = val;
    }
  }
}

// ---------------- fused edge MLP ----------------
// fij = dense2(silu(dense1(concat(s[dst], s[src], d, a, e)))) ; 16 edges / block
#define EKP 296    // 288 + 8 pad
#define HKP 136    // 128 + 8 pad
__global__ void k_edge_mlp(const float* __restrict__ s, const int* __restrict__ ei,
                           const float* __restrict__ dv, const float* __restrict__ av,
                           const float* __restrict__ ef,
                           const __bf16* __restrict__ We0t, const float* __restrict__ be0,
                           const __bf16* __restrict__ We1t, const float* __restrict__ be1,
                           float* __restrict__ fij, int eout) {
  __shared__ __bf16 aij[16][EKP];
  __shared__ __bf16 hid[16][HKP];
  __shared__ int sdst[16], ssrc[16];
  int tid = threadIdx.x;
  int w = tid >> 5, lane = tid & 31;
  int half = lane >> 4, nn = lane & 15;
  int e0 = blockIdx.x * 16;
  if (tid < 16) { sdst[tid] = ei[NEDGES + e0 + tid]; ssrc[tid] = ei[e0 + tid]; }
  __syncthreads();
  // phase 0: build 16x288 bf16 input tile (vectorized node-feature gathers)
  {
    int r = tid >> 3, t8 = tid & 7;
    const float4* rowd = (const float4*)(s + sdst[r] * SI);
    const float4* rows = (const float4*)(s + ssrc[r] * SI);
#pragma unroll
    for (int j = 0; j < 4; ++j) {
      int c4 = t8 + j * 8;                    // 32 float4 per 128-wide half
      float4 v1 = rowd[c4];
      unsigned* dp = (unsigned*)&aij[r][c4 << 2];
      dp[0] = pack2bf(v1.x, v1.y); dp[1] = pack2bf(v1.z, v1.w);
      float4 v2 = rows[c4];
      unsigned* dp2 = (unsigned*)&aij[r][SI + (c4 << 2)];
      dp2[0] = pack2bf(v2.x, v2.y); dp2[1] = pack2bf(v2.z, v2.w);
    }
    int eg = e0 + r;
    for (int j = t8; j < EKP - 2 * SI; j += 8) {   // cols 256..295
      float val = 0.f;
      if (j == 0)      val = dv[eg];
      else if (j == 1) val = av[eg];
      else if (j < 2 + EDGE_DIM) val = ef[eg * EDGE_DIM + (j - 2)];
      aij[r][2 * SI + j] = f2bf(val);
    }
  }
  __syncthreads();
  // phase 1: hidden = silu(aij @ We0 + be0); N=128 (32 cols/wave); K=288
  {
    int n0 = w * 32;
    v8f c0 = {}, c1 = {};
    const __bf16* wr0 = We0t + (n0 + nn) * 288 + half * 16;
    const __bf16* wr1 = We0t + (n0 + 16 + nn) * 288 + half * 16;
    for (int ks = 0; ks < 9; ++ks) {
      int k0 = ks * 32;
      v16bf a = load_afrag(&aij[nn][0], k0, half);
      c0 = WMMA_BF16(a, load_bfrag(wr0 + k0), c0);
      c1 = WMMA_BF16(a, load_bfrag(wr1 + k0), c1);
    }
#pragma unroll
    for (int r = 0; r < 8; ++r) {
      int row = half * 8 + r;
      hid[row][n0 + nn]      = f2bf(siluf(c0[r] + be0[n0 + nn]));
      hid[row][n0 + 16 + nn] = f2bf(siluf(c1[r] + be0[n0 + 16 + nn]));
    }
  }
  __syncthreads();
  // phase 2: fij = hid @ We1 + be1; K=128, N=eout (We1t padded to 256 cols)
  int ntiles = (eout + 15) >> 4;
  for (int t = w; t < ntiles; t += 4) {
    int n0 = t * 16;
    v8f c = {};
    const __bf16* wr = We1t + (n0 + nn) * 128 + half * 16;
#pragma unroll
    for (int ks = 0; ks < 4; ++ks) {
      int k0 = ks * 32;
      v16bf a = load_afrag(&hid[nn][0], k0, half);
      c = WMMA_BF16(a, load_bfrag(wr + k0), c);
    }
    int col = n0 + nn;
    if (col < eout) {
      float bv = be1[col];
#pragma unroll
      for (int r = 0; r < 8; ++r) {
        int erow = e0 + half * 8 + r;
        fij[erow * eout + col] = c[r] + bv;
      }
    }
  }
}

// ---------------- segment softmax + scatter ----------------
__global__ void k_edge_max(const float* fij, const int* ei, float* nmax, int eout) {
  int i = blockIdx.x * 256 + threadIdx.x;     // NEDGES*SI
  if (i >= NEDGES * SI) return;
  int e = i >> 7, c = i & 127;
  int d = ei[NEDGES + e];
  atomicMaxF(&nmax[d * SI + c], fij[e * eout + c]);
}
__global__ void k_edge_exp(float* fij, const int* ei, const float* nmax,
                           float* nden, float* deg, int eout) {
  int i = blockIdx.x * 256 + threadIdx.x;
  if (i >= NEDGES * SI) return;
  int e = i >> 7, c = i & 127;
  int d = ei[NEDGES + e];
  float m = nmax[d * SI + c];
  if (m < -3.0e38f) m = 0.f;                  // isfinite guard
  float ev = __expf(fij[e * eout + c] - m);
  fij[e * eout + c] = ev;
  atomicAdd(&nden[d * SI + c], ev);
  if (c == 0) atomicAdd(&deg[d], 1.f);
}
// one wave per edge: attention scatter (128 ch) + gated vector message (3x32)
__global__ void k_edge_scatter(const float* fij, const int* ei, const float* nden,
                               const float* sb, const float* vb, const float* rv,
                               float* s_out, float* v_acc, int eout, int has_v) {
  int e = (blockIdx.x * 256 + threadIdx.x) >> 5;
  int lane = threadIdx.x & 31;
  if (e >= NEDGES) return;
  int src = ei[e], dst = ei[NEDGES + e];
  const float* fe = fij + e * eout;
#pragma unroll
  for (int j = 0; j < 4; ++j) {
    int c = lane + j * 32;
    float att = fe[c] / (nden[dst * SI + c] + EPSF);
    atomicAdd(&s_out[dst * SI + c], att * sb[src * SI + c]);
  }
  float gate = sigmf(fe[SI]);
  float vr = fe[SI + 1 + lane];
  float vvv = has_v ? fe[SI + 1 + VI + lane] : 0.f;
#pragma unroll
  for (int d = 0; d < 3; ++d) {
    float val = rv[e * 3 + d] * vr;
    if (has_v) val += vvv * vb[src * 96 + d * VI + lane];
    atomicAdd(&v_acc[dst * 96 + d * VI + lane], gate * val);
  }
}
__global__ void k_v_mean_update(float* v, const float* v_acc, const float* deg) {
  int i = blockIdx.x * 256 + threadIdx.x;
  if (i >= NNODES * 96) return;
  v[i] += v_acc[i] / fmaxf(deg[i / 96], 1.f);
}

// ---------------- node update ----------------
__global__ void k_sc2(const float* s, const float* vv, float* sc2) {
  int i = blockIdx.x * 256 + threadIdx.x;     // NNODES*160
  if (i >= NNODES * 160) return;
  int n = i / 160, c = i % 160;
  float val;
  if (c < SI) val = s[n * SI + c];
  else {
    int j = c - SI;
    float acc = 0.f;
    for (int d = 0; d < 3; ++d) { float q = vv[(n * 3 + d) * 64 + j]; acc += q * q; }
    val = sqrtf(fmaxf(acc, EPSF));
  }
  sc2[i] = val;
}
__global__ void k_final_update(const float* y, const float* vv, float* s, float* dvtmp) {
  int i = blockIdx.x * 256 + threadIdx.x;     // NNODES*(SI+96)
  if (i >= NNODES * (SI + 96)) return;
  int n = i / (SI + 96), c = i % (SI + 96);
  if (c < SI) {
    s[n * SI + c] += y[n * 160 + c];
  } else {
    int j = c - SI, d = j / VI, cc = j % VI;
    float g = y[n * 160 + SI + cc];
    dvtmp[(n * 3 + d) * VI + cc] = g * vv[(n * 3 + d) * 64 + VI + cc];
  }
}

// ---------------- host orchestration ----------------
extern "C" void kernel_launch(void* const* d_in, const int* in_sizes, int n_in,
                              void* d_out, int out_size, void* d_ws, size_t ws_size,
                              hipStream_t stream) {
  (void)in_sizes; (void)n_in; (void)out_size; (void)ws_size;
  const float* in_s  = (const float*)d_in[0];
  const float* in_v  = (const float*)d_in[1];
  const int*   ei    = (const int*)d_in[3];
  const float* d_loc = (const float*)d_in[4];
  const float* a_loc = (const float*)d_in[5];
  const float* r_loc = (const float*)d_in[6];
  const float* e_loc = (const float*)d_in[7];
  const int*   batch = (const int*)d_in[13];

  struct Layer {
    const float *ln_w, *ln_b, *Ws, *bs, *Wv, *We0, *be0, *We1, *be1, *Wv0;
    const float *Wu0, *bu0, *Wu1, *bu1, *Wv1, *Wu, *bu;
  } L[NLAYERS];
  int pi = 14;
  for (int i = 0; i < NLAYERS; ++i) {
    bool hv = i > 0, mlp = i < NLAYERS - 1;
    L[i].ln_w = (const float*)d_in[pi++];
    L[i].ln_b = (const float*)d_in[pi++];
    L[i].Ws   = (const float*)d_in[pi++];
    L[i].bs   = (const float*)d_in[pi++];
    L[i].Wv   = hv ? (const float*)d_in[pi++] : nullptr;
    L[i].We0  = (const float*)d_in[pi++];
    L[i].be0  = (const float*)d_in[pi++];
    L[i].We1  = (const float*)d_in[pi++];
    L[i].be1  = (const float*)d_in[pi++];
    L[i].Wv0  = (const float*)d_in[pi++];
    if (mlp) {
      L[i].Wu0 = (const float*)d_in[pi++]; L[i].bu0 = (const float*)d_in[pi++];
      L[i].Wu1 = (const float*)d_in[pi++]; L[i].bu1 = (const float*)d_in[pi++];
      L[i].Wv1 = (const float*)d_in[pi++];
      L[i].Wu = nullptr; L[i].bu = nullptr;
    } else {
      L[i].Wu = (const float*)d_in[pi++]; L[i].bu = (const float*)d_in[pi++];
      L[i].Wu0 = L[i].Wu1 = L[i].Wv1 = nullptr; L[i].bu0 = L[i].bu1 = nullptr;
    }
  }

  // workspace carve-up
  char* wp = (char*)d_ws;
  auto carve = [&](size_t bytes) -> void* {
    void* p = (void*)wp; wp += (bytes + 255) & ~(size_t)255; return p;
  };
  float* s_buf  = (float*)carve((size_t)NNODES * SI * 4);
  float* v_buf  = (float*)carve((size_t)NNODES * 96 * 4);
  float* sb     = (float*)carve((size_t)NNODES * SI * 4);
  float* vb     = (float*)carve((size_t)NNODES * 96 * 4);
  float* fij    = (float*)carve((size_t)NEDGES * 193 * 4);
  float* nmax   = (float*)carve((size_t)NNODES * SI * 4);
  float* nden   = (float*)carve((size_t)NNODES * SI * 4);
  float* v_acc  = (float*)carve((size_t)NNODES * 96 * 4);
  float* deg    = (float*)carve((size_t)NNODES * 4);
  float* stats  = (float*)carve((size_t)NGRAPHS * 7 * 4);
  float* gsum_s = stats, *gcnt = stats + 256, *gsum_var = stats + 512,
       * gsum_v = stats + 768, *mu = stats + 1024, *rstd_s = stats + 1280, *rstd_v = stats + 1536;
  float* vv     = (float*)carve((size_t)NNODES * 3 * 64 * 4);
  float* sc2    = (float*)carve((size_t)NNODES * 160 * 4);
  float* u_buf  = (float*)carve((size_t)NNODES * SI * 4);
  float* y_buf  = (float*)carve((size_t)NNODES * 160 * 4);
  float* dvtmp  = (float*)carve((size_t)NNODES * 96 * 4);
  __bf16* Wst  = (__bf16*)carve(128 * 128 * 2);
  __bf16* Wvt  = (__bf16*)carve(64 * 32 * 2);
  __bf16* We0t = (__bf16*)carve(128 * 288 * 2);
  __bf16* We1t = (__bf16*)carve(256 * 128 * 2);
  __bf16* Wv0t = (__bf16*)carve(64 * 32 * 2);
  __bf16* Wu0t = (__bf16*)carve(128 * 160 * 2);
  __bf16* Wu1t = (__bf16*)carve(192 * 128 * 2);
  __bf16* Wut  = (__bf16*)carve(192 * 160 * 2);
  __bf16* Wv1t = (__bf16*)carve(64 * 32 * 2);

  auto blks = [](int n) { return (n + 255) / 256; };
  auto gemm = [&](const float* X, const __bf16* Wt, const float* bias, float* Y,
                  int M, int N, int K, int ldx, int ldy, int Kpad, int act, int addout) {
    dim3 g((N + 63) / 64, M / 16);
    k_gemm<<<g, 128, 0, stream>>>(X, Wt, bias, Y, M, N, K, ldx, ldy, Kpad, act, addout);
  };
  auto convw = [&](const float* W, __bf16* Wt, int K, int N, int Kpad, int Npad) {
    k_convert_wt<<<blks(Kpad * Npad), 256, 0, stream>>>(W, Wt, K, N, Kpad, Npad);
  };

  k_copy<<<blks(NNODES * SI), 256, 0, stream>>>(s_buf, in_s, NNODES * SI);
  k_copy<<<blks(NNODES * 96), 256, 0, stream>>>(v_buf, in_v, NNODES * 96);

  for (int i = 0; i < NLAYERS; ++i) {
    bool hv = i > 0, mlp = i < NLAYERS - 1;
    int eout = SI + 1 + (hv ? 2 : 1) * VI;   // 161 or 193

    // ---- graph layernorms ----
    k_fill<<<blks(NGRAPHS * 7), 256, 0, stream>>>(stats, 0.f, NGRAPHS * 7);
    k_node_stats1<<<NNODES, 128, 0, stream>>>(s_buf, v_buf, batch, gsum_s, gsum_v, gcnt);
    k_graph_mu<<<1, 256, 0, stream>>>(gsum_s, gcnt, mu);
    k_node_stats2<<<NNODES, 128, 0, stream>>>(s_buf, batch, mu, gsum_var);
    k_graph_fin<<<1, 256, 0, stream>>>(gsum_var, gsum_v, gcnt, rstd_s, rstd_v);
    k_normalize<<<blks(NNODES * 224), 256, 0, stream>>>(s_buf, v_buf, batch, mu, rstd_s,
                                                        rstd_v, L[i].ln_w, L[i].ln_b);
    // ---- weight conversion (f32 -> transposed padded bf16) ----
    convw(L[i].Ws, Wst, 128, 128, 128, 128);
    if (hv) convw(L[i].Wv, Wvt, 32, 32, 32, 64);
    convw(L[i].We0, We0t, 274, 128, 288, 128);
    convw(L[i].We1, We1t, 128, eout, 128, 256);
    convw(L[i].Wv0, Wv0t, 32, 64, 32, 64);
    if (mlp) {
      convw(L[i].Wu0, Wu0t, 160, 128, 160, 128);
      convw(L[i].Wu1, Wu1t, 128, 160, 128, 192);
      convw(L[i].Wv1, Wv1t, 32, 32, 32, 64);
    } else {
      convw(L[i].Wu, Wut, 160, 160, 160, 192);
    }
    // ---- node pre-projections ----
    gemm(s_buf, Wst, L[i].bs, sb, NNODES, 128, 128, 128, 128, 128, 0, 0);
    if (hv) gemm(v_buf, Wvt, nullptr, vb, NNODES * 3, 32, 32, 32, 32, 32, 0, 0);
    // ---- fused edge MLP ----
    k_edge_mlp<<<NEDGES / 16, 128, 0, stream>>>(s_buf, ei, d_loc, a_loc, e_loc,
                                                We0t, L[i].be0, We1t, L[i].be1, fij, eout);
    // ---- segment softmax + scatter ----
    k_fill_bits<<<blks(NNODES * SI), 256, 0, stream>>>((unsigned*)nmax, 0xFF800000u, NNODES * SI);
    k_fill<<<blks(NNODES * SI), 256, 0, stream>>>(nden, 0.f, NNODES * SI);
    k_fill<<<blks(NNODES), 256, 0, stream>>>(deg, 0.f, NNODES);
    k_fill<<<blks(NNODES * 96), 256, 0, stream>>>(v_acc, 0.f, NNODES * 96);
    k_edge_max<<<blks(NEDGES * SI), 256, 0, stream>>>(fij, ei, nmax, eout);
    k_edge_exp<<<blks(NEDGES * SI), 256, 0, stream>>>(fij, ei, nmax, nden, deg, eout);
    k_edge_scatter<<<NEDGES * 32 / 256, 256, 0, stream>>>(fij, ei, nden, sb, vb, r_loc,
                                                          s_buf, v_acc, eout, hv ? 1 : 0);
    k_v_mean_update<<<blks(NNODES * 96), 256, 0, stream>>>(v_buf, v_acc, deg);
    // ---- node feedforward ----
    gemm(v_buf, Wv0t, nullptr, vv, NNODES * 3, 64, 32, 32, 64, 32, 0, 0);
    k_sc2<<<blks(NNODES * 160), 256, 0, stream>>>(s_buf, vv, sc2);
    if (mlp) {
      gemm(sc2, Wu0t, L[i].bu0, u_buf, NNODES, 128, 160, 160, 128, 160, 1, 0);
      gemm(u_buf, Wu1t, L[i].bu1, y_buf, NNODES, 160, 128, 128, 160, 128, 0, 0);
    } else {
      gemm(sc2, Wut, L[i].bu, y_buf, NNODES, 160, 160, 160, 160, 160, 0, 0);
    }
    k_final_update<<<blks(NNODES * 224), 256, 0, stream>>>(y_buf, vv, s_buf, dvtmp);
    if (mlp) gemm(dvtmp, Wv1t, nullptr, v_buf, NNODES * 3, 32, 32, 32, 32, 32, 0, 1);
    else     k_add<<<blks(NNODES * 96), 256, 0, stream>>>(v_buf, dvtmp, NNODES * 96);
  }

  float* out = (float*)d_out;
  k_copy<<<blks(NNODES * SI), 256, 0, stream>>>(out, s_buf, NNODES * SI);
  k_copy<<<blks(NNODES * 96), 256, 0, stream>>>(out + (size_t)NNODES * SI, v_buf, NNODES * 96);
}
